// PointConvDensityClsSsg_472446403129
// MI455X (gfx1250) — matrix operand order repaired and checked by reading, not compiled, not run on hardware
//
#include <hip/hip_runtime.h>
#include <hip/hip_bf16.h>
#include <math.h>

// ---------------------------------------------------------------------------
// PointConvDensityClsSsg on MI455X (gfx1250, wave32).
// All GEMM-shaped work (per-point MLPs, agg einsum, linear layers, FC head)
// runs on v_wmma_f32_16x16x32_f16 (f16 in, f32 accumulate), 128x64 block
// tiles, 2x2 16x16 tiles per wave. Irregular work (density, FPS, kNN,
// micro-MLPs) is VALU.
// ---------------------------------------------------------------------------

typedef __attribute__((ext_vector_type(4)))  _Float16 hv4;
typedef __attribute__((ext_vector_type(8)))  _Float16 hv8;
typedef __attribute__((ext_vector_type(16))) _Float16 hv16;
typedef __attribute__((ext_vector_type(8)))  float    fv8;

#define LPAD 8   // LDS row padding (halves) to break bank conflicts

// A fragment (16x32 f16): lane row = l16; halves 0-7 hold K=g16*8..,
// halves 8-15 hold K=16+g16*8.. -> two aligned b128 LDS loads.
__device__ inline hv16 ldfragA(const _Float16* p) {
  hv8 x = *(const hv8*)p;
  hv8 y = *(const hv8*)(p + 16);
  return __builtin_shufflevector(x, y, 0,1,2,3,4,5,6,7,8,9,10,11,12,13,14,15);
}
// B fragment (32x16 f16, staged transposed): lane col = l16; halves 0-7 hold
// K=g16*8.., halves 8-15 hold K=16+g16*8.. -> two aligned b128 LDS loads.
__device__ inline hv16 ldfragB(const _Float16* p) {
  hv8 x = *(const hv8*)p;
  hv8 y = *(const hv8*)(p + 16);
  return __builtin_shufflevector(x, y, 0,1,2,3,4,5,6,7,8,9,10,11,12,13,14,15);
}

// ---------------------------------------------------------------- transpose
__global__ void k_transpose(const float* __restrict__ src, float* __restrict__ dst,
                            int B, int N) {
  int i = blockIdx.x * blockDim.x + threadIdx.x;
  if (i >= B * N) return;
  int b = i / N, n = i % N;
  dst[(size_t)i * 3 + 0] = src[((size_t)b * 3 + 0) * N + n];
  dst[(size_t)i * 3 + 1] = src[((size_t)b * 3 + 1) * N + n];
  dst[(size_t)i * 3 + 2] = src[((size_t)b * 3 + 2) * N + n];
}

// ------------------------------------------------------- density -> 1/dens
// grid: (cdiv(N,256), B). Inner points staged through LDS in 256-pt chunks.
__global__ __launch_bounds__(256) void k_density(const float* __restrict__ xyz,
                                                 float* __restrict__ invd,
                                                 int N, float bw) {
  __shared__ float sx[256], sy[256], sz[256];
  int b = blockIdx.y;
  int i = blockIdx.x * 256 + threadIdx.x;
  bool valid = (i < N);
  const float* base = xyz + (size_t)b * N * 3;
  float px = 0.f, py = 0.f, pz = 0.f;
  if (valid) {
    px = base[(size_t)i * 3 + 0];
    py = base[(size_t)i * 3 + 1];
    pz = base[(size_t)i * 3 + 2];
  }
  const float c = -1.f / (2.f * bw * bw);
  float s = 0.f;
  for (int j0 = 0; j0 < N; j0 += 256) {
    int j = j0 + threadIdx.x;
    if (j < N) {
      sx[threadIdx.x] = base[(size_t)j * 3 + 0];
      sy[threadIdx.x] = base[(size_t)j * 3 + 1];
      sz[threadIdx.x] = base[(size_t)j * 3 + 2];
    }
    __syncthreads();
    int lim = (N - j0 < 256) ? (N - j0) : 256;
    if (valid) {
      for (int j2 = 0; j2 < lim; ++j2) {
        float dx = sx[j2] - px, dy = sy[j2] - py, dz = sz[j2] - pz;
        s += __expf((dx * dx + dy * dy + dz * dz) * c);
      }
    }
    __syncthreads();
  }
  if (valid) {
    float dens = s / (2.5f * bw) / (float)N;   // mean of gaussian kernel
    invd[(size_t)b * N + i] = 1.f / dens;
  }
}

// ------------------------------------------------ farthest point sampling
// One block per batch. Sequential npoint iterations; argmax with lowest-index
// tie-break to match jnp.argmax. fidx[i] = far BEFORE the i-th update (scan
// emits the incoming carry), so fidx[0] == 0.
__global__ __launch_bounds__(256) void k_fps(const float* __restrict__ xyz,
                                             int* __restrict__ fidx,
                                             int N, int npoint) {
  int b = blockIdx.x, t = threadIdx.x;
  __shared__ float dist[2048];
  __shared__ float rv[256];
  __shared__ int   ri[256];
  __shared__ int   sfar;
  for (int j = t; j < N; j += 256) dist[j] = 1e10f;
  if (t == 0) sfar = 0;
  __syncthreads();
  const float* base = xyz + (size_t)b * N * 3;
  for (int i = 0; i < npoint; ++i) {
    int far = sfar;
    if (t == 0) fidx[b * npoint + i] = far;
    float cx = base[far * 3 + 0], cy = base[far * 3 + 1], cz = base[far * 3 + 2];
    float bv = -1.f; int bi = 0;
    for (int j = t; j < N; j += 256) {
      float dx = base[j * 3 + 0] - cx;
      float dy = base[j * 3 + 1] - cy;
      float dz = base[j * 3 + 2] - cz;
      float d  = dx * dx + dy * dy + dz * dz;
      float nd = fminf(dist[j], d);
      dist[j] = nd;
      if (nd > bv) { bv = nd; bi = j; }
    }
    rv[t] = bv; ri[t] = bi;
    __syncthreads();
    for (int off = 128; off > 0; off >>= 1) {
      if (t < off) {
        if (rv[t + off] > rv[t] || (rv[t + off] == rv[t] && ri[t + off] < ri[t])) {
          rv[t] = rv[t + off]; ri[t] = ri[t + off];
        }
      }
      __syncthreads();
    }
    if (t == 0) sfar = ri[0];
    __syncthreads();
  }
}

// --------------------------------------------- gather sampled centroids
__global__ void k_gatherq(const float* __restrict__ xyz, const int* __restrict__ fidx,
                          float* __restrict__ q, int N, int S, int B) {
  int i = blockIdx.x * blockDim.x + threadIdx.x;
  if (i >= B * S) return;
  int b = i / S;
  int g = fidx[i];
  q[(size_t)i * 3 + 0] = xyz[((size_t)b * N + g) * 3 + 0];
  q[(size_t)i * 3 + 1] = xyz[((size_t)b * N + g) * 3 + 1];
  q[(size_t)i * 3 + 2] = xyz[((size_t)b * N + g) * 3 + 2];
}

// ------------------------------------------------------------------- kNN
// One block per (s,b) query: LDS distance table, K repeated argmins (ascending
// order, lowest-index ties == jax.lax.top_k on -dist).
__global__ __launch_bounds__(256) void k_knn(const float* __restrict__ xyz,
                                             const float* __restrict__ q,
                                             int* __restrict__ idx,
                                             int N, int S, int K) {
  int b = blockIdx.y, s = blockIdx.x, t = threadIdx.x;
  __shared__ float d[2048];
  __shared__ float rv[256];
  __shared__ int   ri[256];
  const float* base = xyz + (size_t)b * N * 3;
  float qx = q[((size_t)b * S + s) * 3 + 0];
  float qy = q[((size_t)b * S + s) * 3 + 1];
  float qz = q[((size_t)b * S + s) * 3 + 2];
  for (int j = t; j < N; j += 256) {
    float dx = base[j * 3 + 0] - qx;
    float dy = base[j * 3 + 1] - qy;
    float dz = base[j * 3 + 2] - qz;
    d[j] = dx * dx + dy * dy + dz * dz;
  }
  __syncthreads();
  for (int k = 0; k < K; ++k) {
    float bv = 1e30f; int bi = 0;
    for (int j = t; j < N; j += 256)
      if (d[j] < bv) { bv = d[j]; bi = j; }
    rv[t] = bv; ri[t] = bi;
    __syncthreads();
    for (int off = 128; off > 0; off >>= 1) {
      if (t < off) {
        if (rv[t + off] < rv[t] || (rv[t + off] == rv[t] && ri[t + off] < ri[t])) {
          rv[t] = rv[t + off]; ri[t] = ri[t + off];
        }
      }
      __syncthreads();
    }
    if (t == 0) {
      idx[((size_t)b * S + s) * K + k] = ri[0];
      d[ri[0]] = 1e30f;
    }
    __syncthreads();
  }
}

// ----------------------------------- group: gxyz (centered), gdens, feat
__global__ void k_group(const float* __restrict__ xyz, const float* __restrict__ q,
                        const float* __restrict__ pts, const float* __restrict__ invd,
                        const int* __restrict__ idx,
                        float* __restrict__ gxyz, float* __restrict__ gdens,
                        float* __restrict__ feat,
                        int N, int S, int K, int Cp, int total) {
  int i = blockIdx.x * blockDim.x + threadIdx.x;
  if (i >= total) return;
  int s = (i / K) % S;
  int b = i / (K * S);
  int g = idx[i];
  float gx = xyz[((size_t)b * N + g) * 3 + 0] - q[((size_t)b * S + s) * 3 + 0];
  float gy = xyz[((size_t)b * N + g) * 3 + 1] - q[((size_t)b * S + s) * 3 + 1];
  float gz = xyz[((size_t)b * N + g) * 3 + 2] - q[((size_t)b * S + s) * 3 + 2];
  gxyz[(size_t)i * 3 + 0] = gx;
  gxyz[(size_t)i * 3 + 1] = gy;
  gxyz[(size_t)i * 3 + 2] = gz;
  gdens[i] = invd[(size_t)b * N + g];
  if (feat) {
    int Cf = 3 + Cp;
    float* f = feat + (size_t)i * Cf;
    f[0] = gx; f[1] = gy; f[2] = gz;
    const float* p = pts + ((size_t)b * N + g) * Cp;
    for (int c = 0; c < Cp; ++c) f[3 + c] = p[c];
  }
}

// -------------------------------- sa3 group_all: feat = concat(xyz, points)
__global__ void k_feat3(const float* __restrict__ xyz, const float* __restrict__ pts,
                        float* __restrict__ feat, int total, int Cp) {
  int i = blockIdx.x * blockDim.x + threadIdx.x;
  if (i >= total) return;
  int Cf = 3 + Cp;
  float* f = feat + (size_t)i * Cf;
  f[0] = xyz[(size_t)i * 3 + 0];
  f[1] = xyz[(size_t)i * 3 + 1];
  f[2] = xyz[(size_t)i * 3 + 2];
  const float* p = pts + (size_t)i * Cp;
  for (int c = 0; c < Cp; ++c) f[3 + c] = p[c];
}

// ------------------------------------------ density micro-MLP (1->16->8->1)
// block per (b,s), thread per k. Normalize by per-group max, BN+ReLU, sigmoid.
__global__ void k_density_mlp(const float* __restrict__ gdens,
    const float* __restrict__ W0, const float* __restrict__ b0,
    const float* __restrict__ be0, const float* __restrict__ g0,
    const float* __restrict__ W1, const float* __restrict__ b1,
    const float* __restrict__ be1, const float* __restrict__ g1,
    const float* __restrict__ W2, const float* __restrict__ b2,
    const float* __restrict__ be2, const float* __restrict__ g2,
    float* __restrict__ ds, int K) {
  __shared__ float red[128];
  int bs = blockIdx.x, k = threadIdx.x;
  float v = gdens[(size_t)bs * K + k];
  red[k] = v;
  __syncthreads();
  for (int off = K >> 1; off > 0; off >>= 1) {
    if (k < off) red[k] = fmaxf(red[k], red[k + off]);
    __syncthreads();
  }
  float x = v / red[0];
  const float bns = rsqrtf(1.f + 1e-5f);
  float h1[16];
#pragma unroll
  for (int j = 0; j < 16; ++j)
    h1[j] = fmaxf((x * W0[j] + b0[j]) * (g0[j] * bns) + be0[j], 0.f);
  float h2[8];
#pragma unroll
  for (int d0 = 0; d0 < 8; ++d0) {
    float t = b1[d0];
#pragma unroll
    for (int j = 0; j < 16; ++j) t += h1[j] * W1[j * 8 + d0];
    h2[d0] = fmaxf(t * (g1[d0] * bns) + be1[d0], 0.f);
  }
  float t = b2[0];
#pragma unroll
  for (int d0 = 0; d0 < 8; ++d0) t += h2[d0] * W2[d0];
  t = t * (g2[0] * bns) + be2[0];
  ds[(size_t)bs * K + k] = 1.f / (1.f + __expf(-t));
}

// -------------------------------------------- weight micro-MLP (3->8->8->16)
__global__ void k_weight_mlp(const float* __restrict__ gxyz,
    const float* __restrict__ W0, const float* __restrict__ b0,
    const float* __restrict__ be0, const float* __restrict__ g0,
    const float* __restrict__ W1, const float* __restrict__ b1,
    const float* __restrict__ be1, const float* __restrict__ g1,
    const float* __restrict__ W2, const float* __restrict__ b2,
    const float* __restrict__ be2, const float* __restrict__ g2,
    float* __restrict__ wmat, int total) {
  int i = blockIdx.x * blockDim.x + threadIdx.x;
  if (i >= total) return;
  const float bns = rsqrtf(1.f + 1e-5f);
  float x0 = gxyz[(size_t)i * 3 + 0];
  float x1 = gxyz[(size_t)i * 3 + 1];
  float x2 = gxyz[(size_t)i * 3 + 2];
  float h1[8];
#pragma unroll
  for (int j = 0; j < 8; ++j) {
    float t = x0 * W0[0 * 8 + j] + x1 * W0[1 * 8 + j] + x2 * W0[2 * 8 + j] + b0[j];
    h1[j] = fmaxf(t * (g0[j] * bns) + be0[j], 0.f);
  }
  float h2[8];
#pragma unroll
  for (int d0 = 0; d0 < 8; ++d0) {
    float t = b1[d0];
#pragma unroll
    for (int j = 0; j < 8; ++j) t += h1[j] * W1[j * 8 + d0];
    h2[d0] = fmaxf(t * (g1[d0] * bns) + be1[d0], 0.f);
  }
#pragma unroll
  for (int w = 0; w < 16; ++w) {
    float t = b2[w];
#pragma unroll
    for (int d0 = 0; d0 < 8; ++d0) t += h2[d0] * W2[d0 * 16 + w];
    wmat[(size_t)i * 16 + w] = fmaxf(t * (g2[w] * bns) + be2[w], 0.f);
  }
}

// ---------------------------------------------------------------------------
// Generic fused WMMA GEMM: out = act((A(MxC) * W(CxD) + bias) * g*rsqrt(1+e) + be)
// 256 thr = 8 waves; block tile 128(M) x 64(N); 2x2 16x16 WMMA tiles per wave.
// K staged through LDS in f16 chunks of 32 in exact CDNA5 fragment layout.
// Fast path uses float4 global loads + packed hv4 DS stores; slow path is
// bounds-checked scalar (edge tiles, C in {3,131,259}).
// ---------------------------------------------------------------------------
__global__ __launch_bounds__(256) void k_gemm(const float* __restrict__ A,
    const float* __restrict__ W, const float* __restrict__ bias,
    const float* __restrict__ g, const float* __restrict__ be,
    float* __restrict__ out, int M, int C, int D, int act) {
  __shared__ _Float16 lA[128 * (32 + LPAD)] __attribute__((aligned(16)));
  __shared__ _Float16 lB[64  * (32 + LPAD)] __attribute__((aligned(16)));
  const int ST = 32 + LPAD;
  const int tid = threadIdx.x;
  const int lane = tid & 31, wave = tid >> 5;
  const int l16 = lane & 15, g16 = lane >> 4;
  const int m0 = blockIdx.x * 128, n0 = blockIdx.y * 64;
  const int tm = (wave & 3) * 32, tn = (wave >> 2) * 32;
  const bool fastA = (m0 + 128 <= M) && ((C & 3) == 0);
  const bool fastB = (n0 + 64 <= D) && ((D & 3) == 0);
  fv8 acc[2][2] = {};
  for (int kk = 0; kk < C; kk += 32) {
    const bool fullK = (kk + 32 <= C);
    // ---- stage A tile (128 x 32) as f16 ----
    if (fastA && fullK) {
#pragma unroll
      for (int j = 0; j < 4; ++j) {
        int id = tid + 256 * j;
        int r = id >> 3, cq = (id & 7) * 4;
        const float4 v = *(const float4*)(A + (size_t)(m0 + r) * C + kk + cq);
        hv4 h = { (_Float16)v.x, (_Float16)v.y, (_Float16)v.z, (_Float16)v.w };
        *(hv4*)&lA[r * ST + cq] = h;
      }
      if (kk + 32 < C)   // prefetch next K chunk (global_prefetch_b8)
        __builtin_prefetch(A + (size_t)(m0 + (tid >> 1)) * C + kk + 32, 0, 1);
    } else {
      for (int i = tid; i < 128 * 32; i += 256) {
        int r = i >> 5, c = i & 31;
        int m = m0 + r, k = kk + c;
        float v = (m < M && k < C) ? A[(size_t)m * C + k] : 0.f;
        lA[r * ST + c] = (_Float16)v;
      }
    }
    // ---- stage W tile (32 x 64), transposed lB[n][k] ----
    if (fastB && fullK) {
#pragma unroll
      for (int j = 0; j < 2; ++j) {
        int id = tid + 256 * j;
        int k = id >> 4, n4 = (id & 15) * 4;
        const float4 v = *(const float4*)(W + (size_t)(kk + k) * D + n0 + n4);
        lB[(n4 + 0) * ST + k] = (_Float16)v.x;
        lB[(n4 + 1) * ST + k] = (_Float16)v.y;
        lB[(n4 + 2) * ST + k] = (_Float16)v.z;
        lB[(n4 + 3) * ST + k] = (_Float16)v.w;
      }
    } else {
      for (int i = tid; i < 64 * 32; i += 256) {
        int n = i >> 5, c = i & 31;
        int k = kk + c, nn = n0 + n;
        float v = (k < C && nn < D) ? W[(size_t)k * D + nn] : 0.f;
        lB[n * ST + c] = (_Float16)v;
      }
    }
    __syncthreads();
    const _Float16* pa = &lA[(tm + l16) * ST + g16 * 8];
    hv16 a0 = ldfragA(pa);
    hv16 a1 = ldfragA(pa + 16 * ST);
    const _Float16* pb = &lB[(tn + l16) * ST + g16 * 8];
    hv16 b0 = ldfragB(pb);
    hv16 b1 = ldfragB(pb + 16 * ST);
    acc[0][0] = __builtin_amdgcn_wmma_f32_16x16x32_f16(false, a0, false, b0, (short)0, acc[0][0], false, false);
    acc[0][1] = __builtin_amdgcn_wmma_f32_16x16x32_f16(false, a0, false, b1, (short)0, acc[0][1], false, false);
    acc[1][0] = __builtin_amdgcn_wmma_f32_16x16x32_f16(false, a1, false, b0, (short)0, acc[1][0], false, false);
    acc[1][1] = __builtin_amdgcn_wmma_f32_16x16x32_f16(false, a1, false, b1, (short)0, acc[1][1], false, false);
    __syncthreads();
  }
  // epilogue: D VGPR r -> M = r (lanes 0-15) / 8+r (lanes 16-31), N = l16
  const float bns = rsqrtf(1.f + 1e-5f);
#pragma unroll
  for (int tj = 0; tj < 2; ++tj) {
    int n = n0 + tn + tj * 16 + l16;
    if (n < D) {
      float sc = g ? g[n] * bns : 1.f;
      float sh = be ? be[n] : 0.f;
      float bb = bias ? bias[n] : 0.f;
#pragma unroll
      for (int ti = 0; ti < 2; ++ti) {
#pragma unroll
        for (int r = 0; r < 8; ++r) {
          int m = m0 + tm + ti * 16 + r + g16 * 8;
          if (m < M) {
            float v = (acc[ti][tj][r] + bb) * sc + sh;
            if (act == 1) v = fmaxf(v, 0.f);
            else if (act == 2) v = 1.f / (1.f + __expf(-v));
            out[(size_t)m * D + n] = v;
          }
        }
      }
    }
  }
}

// ---------------------------------------------------------------------------
// agg einsum 'bskc,bskw->bscw' as WMMA: per (b,s) GEMM (C x K)*(K x 16).
// grid: (B*S, C/128). h (gated by ds) staged transposed into LDS with
// coalesced float4 global reads; one wave per 16-channel tile; K in {32,64,128}.
// ---------------------------------------------------------------------------
__global__ __launch_bounds__(256) void k_agg(const float* __restrict__ h,
    const float* __restrict__ ds, const float* __restrict__ wm,
    float* __restrict__ agg, int K, int C) {
  __shared__ _Float16 lHT[128 * (128 + LPAD)] __attribute__((aligned(16)));
  __shared__ _Float16 lWT[16  * (128 + LPAD)] __attribute__((aligned(16)));
  const int tid = threadIdx.x;
  const int lane = tid & 31, wave = tid >> 5;
  const int l16 = lane & 15, g16 = lane >> 4;
  const int bs = blockIdx.x, cc = blockIdx.y;
  const int stride = K + LPAD;
  // hT[c][k] = h[bs,k, cc*128+c] * ds[bs,k]; float4 along c (coalesced)
  for (int id = tid; id < K * 32; id += 256) {
    int k = id >> 5, c4 = (id & 31) * 4;
    const float4 v = *(const float4*)(h + ((size_t)bs * K + k) * C + cc * 128 + c4);
    float dv = ds[(size_t)bs * K + k];
    lHT[(c4 + 0) * stride + k] = (_Float16)(v.x * dv);
    lHT[(c4 + 1) * stride + k] = (_Float16)(v.y * dv);
    lHT[(c4 + 2) * stride + k] = (_Float16)(v.z * dv);
    lHT[(c4 + 3) * stride + k] = (_Float16)(v.w * dv);
  }
  // wT[w][k]; float4 along w (coalesced)
  for (int id = tid; id < K * 4; id += 256) {
    int k = id >> 2, w4 = (id & 3) * 4;
    const float4 v = *(const float4*)(wm + ((size_t)bs * K + k) * 16 + w4);
    lWT[(w4 + 0) * stride + k] = (_Float16)v.x;
    lWT[(w4 + 1) * stride + k] = (_Float16)v.y;
    lWT[(w4 + 2) * stride + k] = (_Float16)v.z;
    lWT[(w4 + 3) * stride + k] = (_Float16)v.w;
  }
  __syncthreads();
  fv8 acc = {};
  for (int kk = 0; kk < K; kk += 32) {
    hv16 af = ldfragA(&lHT[(wave * 16 + l16) * stride + kk + g16 * 8]);
    hv16 bf = ldfragB(&lWT[l16 * stride + kk + g16 * 8]);
    acc = __builtin_amdgcn_wmma_f32_16x16x32_f16(false, af, false, bf,
                                                 (short)0, acc, false, false);
  }
#pragma unroll
  for (int r = 0; r < 8; ++r) {
    int c = cc * 128 + wave * 16 + r + g16 * 8;
    agg[((size_t)bs * C + c) * 16 + l16] = acc[r];
  }
}

// ---------------------------------------------------------------------------
// Host side
// ---------------------------------------------------------------------------
struct L4 { const float *W, *b, *be, *g; };
struct SAp {
  const float *blbe, *blg;
  L4 den[3];
  const float *linW, *linb;
  L4 mlp[3];
  L4 wt[3];
};

extern "C" void kernel_launch(void* const* d_in, const int* in_sizes, int n_in,
                              void* d_out, int out_size, void* d_ws, size_t ws_size,
                              hipStream_t stream) {
  (void)in_sizes; (void)n_in; (void)out_size; (void)ws_size;
  // ---- decode params (JAX pytree flatten: dict keys sorted at every level) --
  int pi = 0;
  const float* xyz = (const float*)d_in[pi++];
  auto F = [&]() { return (const float*)d_in[pi++]; };
  const float *bn1be = F(), *bn1g = F(), *bn2be = F(), *bn2g = F();
  const float *fc1W = F(), *fc1b = F(), *fc2W = F(), *fc2b = F();
  const float *fc3W = F(), *fc3b = F();
  auto LY = [&]() { L4 l; l.W = F(); l.b = F(); l.be = F(); l.g = F(); return l; };
  SAp sa[3];
  for (int s = 0; s < 3; ++s) {
    sa[s].blbe = F(); sa[s].blg = F();
    for (int i = 0; i < 3; ++i) sa[s].den[i] = LY();
    sa[s].linW = F(); sa[s].linb = F();
    for (int i = 0; i < 3; ++i) sa[s].mlp[i] = LY();
    for (int i = 0; i < 3; ++i) sa[s].wt[i]  = LY();
  }

  const int B = 16;
  const int N1 = 2048, S1 = 512, K1 = 32;
  const int N2 = 512,  S2 = 128, K2 = 64;
  const int N3 = 128,  K3 = 128;   // group_all, S3 = 1

  // ---- workspace carve ----------------------------------------------------
  char* base = (char*)d_ws;
  size_t off = 0;
  auto allocf = [&](size_t n) { float* p = (float*)(base + off);
                                off += ((n * 4 + 255) / 256) * 256; return p; };
  auto alloci = [&](size_t n) { int* p = (int*)(base + off);
                                off += ((n * 4 + 255) / 256) * 256; return p; };
  float* x0    = allocf((size_t)B * N1 * 3);
  float* invd1 = allocf((size_t)B * N1);
  int*   fidx1 = alloci((size_t)B * S1);
  float* nx1   = allocf((size_t)B * S1 * 3);
  int*   idx1  = alloci((size_t)B * S1 * K1);
  float* gxyz1 = allocf((size_t)B * S1 * K1 * 3);
  float* gdens1= allocf((size_t)B * S1 * K1);
  float* ds1   = allocf((size_t)B * S1 * K1);
  float* wmat1 = allocf((size_t)B * S1 * K1 * 16);
  float* out1  = allocf((size_t)B * S1 * 128);
  float* invd2 = allocf((size_t)B * N2);
  int*   fidx2 = alloci((size_t)B * S2);
  float* nx2   = allocf((size_t)B * S2 * 3);
  int*   idx2  = alloci((size_t)B * S2 * K2);
  float* gxyz2 = allocf((size_t)B * S2 * K2 * 3);
  float* gdens2= allocf((size_t)B * S2 * K2);
  float* ds2   = allocf((size_t)B * S2 * K2);
  float* wmat2 = allocf((size_t)B * S2 * K2 * 16);
  float* out2  = allocf((size_t)B * S2 * 256);
  float* invd3 = allocf((size_t)B * N3);
  float* ds3   = allocf((size_t)B * K3);
  float* wmat3 = allocf((size_t)B * K3 * 16);
  float* agg3  = allocf((size_t)B * 1024 * 16);
  float* out3  = allocf((size_t)B * 1024);
  float* fcx1  = allocf((size_t)B * 512);
  float* fcx2  = allocf((size_t)B * 256);
  float* H1    = allocf((size_t)33554432);   // mlp ping (max 262144 x 128)
  float* H2    = allocf((size_t)33554432);   // mlp pong
  float* CBUF  = allocf((size_t)17170432);   // feat / agg arena

  auto cdiv = [](int a, int b) { return (a + b - 1) / b; };
  auto gemm = [&](const float* A, const float* W, const float* b,
                  const float* g, const float* be, float* O,
                  int M, int C, int D, int act) {
    dim3 gr(cdiv(M, 128), cdiv(D, 64));
    k_gemm<<<gr, 256, 0, stream>>>(A, W, b, g, be, O, M, C, D, act);
  };
  auto dmlp = [&](const float* gd, const SAp& p, float* dso, int BS, int K) {
    k_density_mlp<<<BS, K, 0, stream>>>(gd,
      p.den[0].W, p.den[0].b, p.den[0].be, p.den[0].g,
      p.den[1].W, p.den[1].b, p.den[1].be, p.den[1].g,
      p.den[2].W, p.den[2].b, p.den[2].be, p.den[2].g, dso, K);
  };
  auto wmlp = [&](const float* gx, const SAp& p, float* wo, int total) {
    k_weight_mlp<<<cdiv(total, 256), 256, 0, stream>>>(gx,
      p.wt[0].W, p.wt[0].b, p.wt[0].be, p.wt[0].g,
      p.wt[1].W, p.wt[1].b, p.wt[1].be, p.wt[1].g,
      p.wt[2].W, p.wt[2].b, p.wt[2].be, p.wt[2].g, wo, total);
  };

  // ======================= SA1 (N=2048 -> S=512, K=32) ======================
  k_transpose<<<cdiv(B * N1, 256), 256, 0, stream>>>(xyz, x0, B, N1);
  k_density<<<dim3(cdiv(N1, 256), B), 256, 0, stream>>>(x0, invd1, N1, 0.1f);
  k_fps<<<B, 256, 0, stream>>>(x0, fidx1, N1, S1);
  k_gatherq<<<cdiv(B * S1, 256), 256, 0, stream>>>(x0, fidx1, nx1, N1, S1, B);
  k_knn<<<dim3(S1, B), 256, 0, stream>>>(x0, nx1, idx1, N1, S1, K1);
  int tot1 = B * S1 * K1;
  k_group<<<cdiv(tot1, 256), 256, 0, stream>>>(x0, nx1, nullptr, invd1, idx1,
      gxyz1, gdens1, nullptr, N1, S1, K1, 0, tot1);
  // mlp: 3 -> 64 -> 64 -> 128 (feat == gxyz for sa1)
  gemm(gxyz1, sa[0].mlp[0].W, sa[0].mlp[0].b, sa[0].mlp[0].g, sa[0].mlp[0].be, H1, tot1, 3,  64, 1);
  gemm(H1,    sa[0].mlp[1].W, sa[0].mlp[1].b, sa[0].mlp[1].g, sa[0].mlp[1].be, H2, tot1, 64, 64, 1);
  gemm(H2,    sa[0].mlp[2].W, sa[0].mlp[2].b, sa[0].mlp[2].g, sa[0].mlp[2].be, H1, tot1, 64, 128, 1);
  dmlp(gdens1, sa[0], ds1, B * S1, K1);
  wmlp(gxyz1, sa[0], wmat1, tot1);
  k_agg<<<dim3(B * S1, 1), 256, 0, stream>>>(H1, ds1, wmat1, CBUF, K1, 128);
  gemm(CBUF, sa[0].linW, sa[0].linb, sa[0].blg, sa[0].blbe, out1, B * S1, 16 * 128, 128, 1);

  // ======================= SA2 (N=512 -> S=128, K=64) =======================
  k_density<<<dim3(cdiv(N2, 256), B), 256, 0, stream>>>(nx1, invd2, N2, 0.2f);
  k_fps<<<B, 256, 0, stream>>>(nx1, fidx2, N2, S2);
  k_gatherq<<<cdiv(B * S2, 256), 256, 0, stream>>>(nx1, fidx2, nx2, N2, S2, B);
  k_knn<<<dim3(S2, B), 256, 0, stream>>>(nx1, nx2, idx2, N2, S2, K2);
  int tot2 = B * S2 * K2;
  k_group<<<cdiv(tot2, 256), 256, 0, stream>>>(nx1, nx2, out1, invd2, idx2,
      gxyz2, gdens2, CBUF, N2, S2, K2, 128, tot2);   // feat: (tot2 x 131)
  // mlp: 131 -> 128 -> 128 -> 256
  gemm(CBUF, sa[1].mlp[0].W, sa[1].mlp[0].b, sa[1].mlp[0].g, sa[1].mlp[0].be, H1, tot2, 131, 128, 1);
  gemm(H1,   sa[1].mlp[1].W, sa[1].mlp[1].b, sa[1].mlp[1].g, sa[1].mlp[1].be, H2, tot2, 128, 128, 1);
  gemm(H2,   sa[1].mlp[2].W, sa[1].mlp[2].b, sa[1].mlp[2].g, sa[1].mlp[2].be, H1, tot2, 128, 256, 1);
  dmlp(gdens2, sa[1], ds2, B * S2, K2);
  wmlp(gxyz2, sa[1], wmat2, tot2);
  k_agg<<<dim3(B * S2, 2), 256, 0, stream>>>(H1, ds2, wmat2, CBUF, K2, 256);
  gemm(CBUF, sa[1].linW, sa[1].linb, sa[1].blg, sa[1].blbe, out2, B * S2, 16 * 256, 256, 1);

  // ======================= SA3 (group_all, K=128) ===========================
  k_density<<<dim3(cdiv(N3, 256), B), 256, 0, stream>>>(nx2, invd3, N3, 0.4f);
  int tot3 = B * K3;
  k_feat3<<<cdiv(tot3, 256), 256, 0, stream>>>(nx2, out2, CBUF, tot3, 256); // (tot3 x 259)
  // mlp: 259 -> 256 -> 512 -> 1024
  gemm(CBUF, sa[2].mlp[0].W, sa[2].mlp[0].b, sa[2].mlp[0].g, sa[2].mlp[0].be, H1, tot3, 259, 256, 1);
  gemm(H1,   sa[2].mlp[1].W, sa[2].mlp[1].b, sa[2].mlp[1].g, sa[2].mlp[1].be, H2, tot3, 256, 512, 1);
  gemm(H2,   sa[2].mlp[2].W, sa[2].mlp[2].b, sa[2].mlp[2].g, sa[2].mlp[2].be, H1, tot3, 512, 1024, 1);
  dmlp(invd3, sa[2], ds3, B, K3);           // gdens == inv density, group_all
  wmlp(nx2, sa[2], wmat3, tot3);            // weight-mlp on raw (uncentered) xyz
  k_agg<<<dim3(B, 8), 256, 0, stream>>>(H1, ds3, wmat3, agg3, K3, 1024);
  gemm(agg3, sa[2].linW, sa[2].linb, sa[2].blg, sa[2].blbe, out3, B, 16 * 1024, 1024, 1);

  // ======================= FC head =========================================
  gemm(out3, fc1W, fc1b, bn1g, bn1be, fcx1, B, 1024, 512, 1);
  gemm(fcx1, fc2W, fc2b, bn2g, bn2be, fcx2, B, 512, 256, 1);
  gemm(fcx2, fc3W, fc3b, nullptr, nullptr, (float*)d_out, B, 256, 40, 0);
}
